// PCS_Mviews_CrossModel_47545287967134
// MI455X (gfx1250) — matrix operand order repaired and checked by reading, not compiled, not run on hardware
//
#include <hip/hip_runtime.h>
#include <hip/hip_bf16.h>

typedef __attribute__((ext_vector_type(16))) _Float16 v16h;
typedef __attribute__((ext_vector_type(8)))  _Float16 v8h;
typedef __attribute__((ext_vector_type(8)))  float    v8f;

#define EPSBN 1e-5f
#define KNN   20
#define B2    16
#define NPTS  1024
#define NEGINF (-3.4e38f)

__device__ __forceinline__ v8f wmma32f16(v16h a, v16h b, v8f c) {
  // D = A(16x32,f16) * B(32x16,f16) + C(16x16,f32)
  return __builtin_amdgcn_wmma_f32_16x16x32_f16(false, a, false, b, (short)0, c, false, false);
}

// Per-lane A fragment from f16 row-major memory (row = lane&15; K split per ISA layout)
__device__ __forceinline__ v16h load_a_f16(const _Float16* __restrict__ ap) {
  v8h lo = *(const v8h*)ap;          // K = kk + gl*8 + 0..7
  v8h hi = *(const v8h*)(ap + 16);   // K = kk + 16 + gl*8 + 0..7
  v16h a;
  #pragma unroll
  for (int i = 0; i < 8; ++i) { a[i] = lo[i]; a[i + 8] = hi[i]; }
  return a;
}

// ---------------------------------------------------------------- utilities
__global__ void build_pts(const float* __restrict__ x, const float* __restrict__ x2,
                          float* __restrict__ pts) {
  int i = blockIdx.x * blockDim.x + threadIdx.x;
  const int tot = B2 * NPTS * 3;
  if (i >= tot) return;
  const int half = 8 * NPTS * 3;
  pts[i] = (i < half) ? x[i] : x2[i - half];
}

__global__ void cvt_f32_f16(const float* __restrict__ w, _Float16* __restrict__ o,
                            int O, int Kin, int Kp) {
  int i = blockIdx.x * blockDim.x + threadIdx.x;
  if (i >= O * Kp) return;
  int r = i / Kp, c = i - r * Kp;
  o[i] = (c < Kin) ? (_Float16)w[r * Kin + c] : (_Float16)0.f;
}

// ---------------------------------------------------------------- kNN (fp32, LDS argmax)
__global__ void knn_kernel(const float* __restrict__ h, int stride, int C,
                           int* __restrict__ idxout) {
  __shared__ float ctr[256];
  __shared__ float dist[NPTS];
  __shared__ float redv[128];
  __shared__ int   redi[128];
  const int bn  = blockIdx.x;           // b*NPTS + n
  const int b   = bn / NPTS;
  const int tid = threadIdx.x;          // 128 threads
  const float* hb = h + (size_t)b * NPTS * stride;
  const float* hc = h + (size_t)bn * stride;
  for (int c = tid; c < C; c += 128) ctr[c] = hc[c];
  __syncthreads();
  for (int m = tid; m < NPTS; m += 128) {
    const float* hm = hb + (size_t)m * stride;
    if (m + 128 < NPTS) __builtin_prefetch(hm + (size_t)128 * stride, 0, 0);
    float d = 0.f;
    for (int c = 0; c < C; ++c) { float df = hm[c] - ctr[c]; d -= df * df; }
    dist[m] = d;                        // neg squared distance
  }
  __syncthreads();
  for (int j = 0; j < KNN; ++j) {
    float bv = NEGINF; int bi = NPTS;
    for (int m = tid; m < NPTS; m += 128) {
      float v = dist[m];
      if (v > bv || (v == bv && m < bi)) { bv = v; bi = m; }
    }
    redv[tid] = bv; redi[tid] = bi;
    __syncthreads();
    for (int s = 64; s > 0; s >>= 1) {
      if (tid < s) {
        float ov = redv[tid + s]; int oi = redi[tid + s];
        if (ov > redv[tid] || (ov == redv[tid] && oi < redi[tid])) {
          redv[tid] = ov; redi[tid] = oi;
        }
      }
      __syncthreads();
    }
    if (tid == 0) { idxout[(size_t)bn * KNN + j] = redi[0]; dist[redi[0]] = NEGINF; }
    __syncthreads();
  }
}

// ------------------------------------------------- fused edge-conv (WMMA + BN/leaky + max_k)
// A rows = 20 edge rows (padded to 32 -> two 16-row M tiles), K = 2C padded to Kp, N = O.
// Writes f32 (kNN input for next layer) and f16 (GEMM A operand) copies of the output.
template<int C, int Kp, int O, int NW>
__global__ __launch_bounds__(NW * 32)
void edge_conv_kernel(const float* __restrict__ h, int stride,
                      const int* __restrict__ idx,
                      const _Float16* __restrict__ wf,   // (O, Kp) f16, zero padded
                      const float* __restrict__ bnp,     // (4, O): g,b,m,v
                      float* __restrict__ out, _Float16* __restrict__ outh, int ostride) {
  __shared__ _Float16 e[32 * Kp];
  const int bn  = blockIdx.x;
  const int b   = bn / NPTS;
  const int tid = threadIdx.x;
  const float* hb = h + (size_t)b * NPTS * stride;
  const float* hc = h + (size_t)bn * stride;
  const int*   ix = idx + (size_t)bn * KNN;
  const int C2 = 2 * C;
  for (int i = tid; i < 32 * Kp; i += NW * 32) {
    int r = i / Kp, c = i - r * Kp;
    float v = 0.f;
    if (r < KNN && c < C2) {
      if (c < C) { int nb = ix[r]; v = hb[(size_t)nb * stride + c] - hc[c]; }
      else       { v = hc[c - C]; }
    }
    e[i] = (_Float16)v;
  }
  __syncthreads();

  const int wave = tid >> 5;
  const int lane = tid & 31;
  const int gl   = lane >> 4;      // half-wave group
  const int col  = lane & 15;

  for (int t = wave; t < O / 16; t += NW) {
    const int o0 = t * 16;
    v8f acc0 = {}; v8f acc1 = {};
    for (int kk = 0; kk < Kp; kk += 32) {
      // B: lane holds column o0+col, K = kk + gl*8 + 0..7 and kk + 16 + gl*8 + 0..7
      v16h bv = load_a_f16(wf + (size_t)(o0 + col) * Kp + kk + gl * 8);
      v16h a0 = load_a_f16(e + col * Kp + kk + gl * 8);
      v16h a1 = load_a_f16(e + (16 + col) * Kp + kk + gl * 8);
      acc0 = wmma32f16(a0, bv, acc0);
      acc1 = wmma32f16(a1, bv, acc1);
    }
    const int o = o0 + col;
    const float gg = bnp[o], bb = bnp[O + o], mm = bnp[2 * O + o], vv = bnp[3 * O + o];
    const float sc = gg * rsqrtf(vv + EPSBN);
    float mx = NEGINF;
    #pragma unroll
    for (int gi = 0; gi < 8; ++gi) {            // tile0 rows: k = gi + gl*8 (all valid)
      float y = (acc0[gi] - mm) * sc + bb;
      y = (y >= 0.f) ? y : 0.2f * y;
      mx = fmaxf(mx, y);
    }
    if (gl == 0) {                               // tile1 rows: k = 16+gi, valid gi<4
      #pragma unroll
      for (int gi = 0; gi < 4; ++gi) {
        float y = (acc1[gi] - mm) * sc + bb;
        y = (y >= 0.f) ? y : 0.2f * y;
        mx = fmaxf(mx, y);
      }
    }
    float other = __shfl_xor(mx, 16, 32);        // combine lane L <-> L+16 (same column)
    mx = fmaxf(mx, other);
    if (gl == 0) {
      out [(size_t)bn * ostride + o] = mx;
      outh[(size_t)bn * ostride + o] = (_Float16)mx;
    }
  }
}

// ------------------------------------------------- generic dense GEMM (WMMA) + epilogue
// out[M,N] = act( bn( A[M,K](f16) @ Bw[N,K]^T + bias ) )
// grid (M/(16*MT), N/16), block 32; MT M-tiles per wave reuse each B fragment.
template<int MT>
__global__ __launch_bounds__(32)
void gemm_wmma(const _Float16* __restrict__ A, int lda,
               const _Float16* __restrict__ Bw, int ldb,
               const float* __restrict__ bias, const float* __restrict__ bnp,
               int Ntotal, int K, int act,
               float* __restrict__ out, int ldo, _Float16* __restrict__ outh) {
  const int mt0 = blockIdx.x * MT, nt = blockIdx.y;
  const int lane = threadIdx.x;
  const int gl = lane >> 4, col = lane & 15;
  const int oc = nt * 16 + col;
  v8f acc[MT];
  #pragma unroll
  for (int j = 0; j < MT; ++j) acc[j] = (v8f){};
  for (int kk = 0; kk < K; kk += 32) {
    v16h bv = load_a_f16(Bw + (size_t)oc * ldb + kk + gl * 8);
    #pragma unroll
    for (int j = 0; j < MT; ++j) {
      const int row = (mt0 + j) * 16 + col;
      v16h av = load_a_f16(A + (size_t)row * lda + kk + gl * 8);
      acc[j] = wmma32f16(av, bv, acc[j]);
    }
  }
  float sc = 1.f, sh = 0.f;
  if (bnp) {
    float gg = bnp[oc], bb = bnp[Ntotal + oc], mm = bnp[2 * Ntotal + oc], vv = bnp[3 * Ntotal + oc];
    sc = gg * rsqrtf(vv + EPSBN);
    sh = bb - mm * sc;
  }
  const float bs = bias ? bias[oc] : 0.f;
  #pragma unroll
  for (int j = 0; j < MT; ++j) {
    #pragma unroll
    for (int gi = 0; gi < 8; ++gi) {
      int r = (mt0 + j) * 16 + gi + gl * 8;
      float y = (acc[j][gi] + bs) * sc + sh;
      if (act == 1)      y = (y >= 0.f) ? y : 0.2f * y;
      else if (act == 2) y = fmaxf(y, 0.f);
      out[(size_t)r * ldo + oc] = y;
      if (outh) outh[(size_t)r * ldo + oc] = (_Float16)y;
    }
  }
}

// ------------------------------------------------- reductions & adapters
__global__ void reduce_h5(const float* __restrict__ h5,
                          float* __restrict__ feat, _Float16* __restrict__ feath) {
  const int b = blockIdx.x;
  const int o = blockIdx.y * 256 + threadIdx.x;    // < 1024
  const float* p = h5 + (size_t)b * NPTS * 1024 + o;
  float mx = NEGINF, sm = 0.f;
  for (int n = 0; n < NPTS; ++n) { float v = p[(size_t)n * 1024]; mx = fmaxf(mx, v); sm += v; }
  float mean = sm * (1.f / NPTS);
  feat [(size_t)b * 2048 + o]        = mx;
  feat [(size_t)b * 2048 + 1024 + o] = mean;
  feath[(size_t)b * 2048 + o]        = (_Float16)mx;
  feath[(size_t)b * 2048 + 1024 + o] = (_Float16)mean;
}

__global__ void img_mean(const float* __restrict__ im, float* __restrict__ o) {
  int i = blockIdx.x * blockDim.x + threadIdx.x;   // 8*512
  if (i >= 8 * 512) return;
  int b = i / 512, c = i - b * 512;
  float s = 0.f;
  for (int v = 0; v < 10; ++v) s += im[(size_t)(b * 10 + v) * 512 + c];
  o[i] = s * 0.1f;
}

__global__ void adapter_kernel(const float* __restrict__ pf, const float* __restrict__ imgm,
                               const float* __restrict__ dw, const float* __restrict__ db,
                               const float* __restrict__ uw, const float* __restrict__ ub,
                               float* __restrict__ out) {
  __shared__ float z[512];
  __shared__ float t[256];
  const int blk = blockIdx.x;          // 0..23  (which*8 + row)
  const int which = blk / 8, r = blk - which * 8;
  const float* zp = (which == 0) ? pf + (size_t)r * 512
                   : (which == 1) ? pf + (size_t)(8 + r) * 512
                                  : imgm + (size_t)r * 512;
  const int tid = threadIdx.x;         // 256
  for (int c = tid; c < 512; c += 256) z[c] = zp[c];
  __syncthreads();
  float acc = db[tid];
  for (int c = 0; c < 512; ++c) acc += z[c] * dw[(size_t)tid * 512 + c];
  t[tid] = fmaxf(acc, 0.f);
  __syncthreads();
  for (int o = tid; o < 512; o += 256) {
    float a2 = ub[o];
    for (int c = 0; c < 256; ++c) a2 += t[c] * uw[(size_t)o * 256 + c];
    out[(size_t)blk * 512 + o] = a2;
  }
}

// ---------------------------------------------------------------- launch
extern "C" void kernel_launch(void* const* d_in, const int* in_sizes, int n_in,
                              void* d_out, int out_size, void* d_ws, size_t ws_size,
                              hipStream_t stream) {
  (void)in_sizes; (void)n_in; (void)out_size; (void)ws_size;
  const float* x      = (const float*)d_in[0];
  const float* x2     = (const float*)d_in[1];
  const float* xmv    = (const float*)d_in[2];
  const float* w1     = (const float*)d_in[3];
  const float* bn1    = (const float*)d_in[4];
  const float* w2     = (const float*)d_in[5];
  const float* bn2    = (const float*)d_in[6];
  const float* w3     = (const float*)d_in[7];
  const float* bn3    = (const float*)d_in[8];
  const float* w4     = (const float*)d_in[9];
  const float* bn4    = (const float*)d_in[10];
  const float* w5     = (const float*)d_in[11];
  const float* bn5    = (const float*)d_in[12];
  const float* ih_w1  = (const float*)d_in[13];
  const float* ih_b1  = (const float*)d_in[14];
  const float* ih_bn  = (const float*)d_in[15];
  const float* ih_w2  = (const float*)d_in[16];
  const float* ih_b2  = (const float*)d_in[17];
  const float* img_w1 = (const float*)d_in[18];
  const float* img_bn = (const float*)d_in[19];
  const float* img_w2 = (const float*)d_in[20];
  const float* ad_dw  = (const float*)d_in[21];
  const float* ad_db  = (const float*)d_in[22];
  const float* ad_uw  = (const float*)d_in[23];
  const float* ad_ub  = (const float*)d_in[24];
  float* out = (float*)d_out;

  char* wsb = (char*)d_ws;
  size_t off = 0;
  auto alloc = [&](size_t bytes) -> void* {
    void* p = wsb + off;
    off = (off + bytes + 255) & ~(size_t)255;
    return p;
  };
  float*    pts   = (float*)alloc((size_t)B2 * NPTS * 3 * 4);
  float*    H     = (float*)alloc((size_t)B2 * NPTS * 512 * 4);   // f32 (kNN inputs)
  _Float16* Hh    = (_Float16*)alloc((size_t)B2 * NPTS * 512 * 2); // f16 (GEMM A)
  float*    h5b   = (float*)alloc((size_t)B2 * NPTS * 1024 * 4);
  int*      idxb  = (int*)  alloc((size_t)B2 * NPTS * KNN * 4);
  float*    feat  = (float*)alloc((size_t)B2 * 2048 * 4);
  _Float16* feath = (_Float16*)alloc((size_t)B2 * 2048 * 2);
  float*    t_ih  = (float*)alloc((size_t)B2 * 1024 * 4);
  _Float16* t_h   = (_Float16*)alloc((size_t)B2 * 1024 * 2);
  float*    pf    = (float*)alloc((size_t)B2 * 512 * 4);
  float*    img1  = (float*)alloc((size_t)80 * 512 * 4);
  _Float16* img1h = (_Float16*)alloc((size_t)80 * 512 * 2);
  float*    img2  = (float*)alloc((size_t)80 * 512 * 4);
  float*    imgm  = (float*)alloc((size_t)8 * 512 * 4);
  _Float16* xmvh  = (_Float16*)alloc((size_t)80 * 2048 * 2);
  _Float16* w1p   = (_Float16*)alloc((size_t)64 * 32 * 2);
  _Float16* w2p   = (_Float16*)alloc((size_t)64 * 128 * 2);
  _Float16* w3p   = (_Float16*)alloc((size_t)128 * 256 * 2);
  _Float16* w4p   = (_Float16*)alloc((size_t)256 * 512 * 2);
  _Float16* w5p   = (_Float16*)alloc((size_t)1024 * 512 * 2);
  _Float16* ihw1p = (_Float16*)alloc((size_t)1024 * 2048 * 2);
  _Float16* ihw2p = (_Float16*)alloc((size_t)512 * 1024 * 2);
  _Float16* imw1p = (_Float16*)alloc((size_t)512 * 2048 * 2);
  _Float16* imw2p = (_Float16*)alloc((size_t)512 * 512 * 2);

  build_pts<<<(B2 * NPTS * 3 + 255) / 256, 256, 0, stream>>>(x, x2, pts);

  auto cvt = [&](const float* src, _Float16* dst, int O, int Kin, int Kp) {
    cvt_f32_f16<<<(O * Kp + 255) / 256, 256, 0, stream>>>(src, dst, O, Kin, Kp);
  };
  cvt(w1, w1p, 64, 6, 32);
  cvt(w2, w2p, 64, 128, 128);
  cvt(w3, w3p, 128, 256, 256);
  cvt(w4, w4p, 256, 512, 512);
  cvt(w5, w5p, 1024, 512, 512);
  cvt(ih_w1, ihw1p, 1024, 2048, 2048);
  cvt(ih_w2, ihw2p, 512, 1024, 1024);
  cvt(img_w1, imw1p, 512, 2048, 2048);
  cvt(img_w2, imw2p, 512, 512, 512);
  cvt(xmv, xmvh, 80, 2048, 2048);

  // ---- edge conv stack (h1..h4 packed into H/Hh at channel offsets 0/64/128/256)
  knn_kernel<<<B2 * NPTS, 128, 0, stream>>>(pts, 3, 3, idxb);
  edge_conv_kernel<3, 32, 64, 4><<<B2 * NPTS, 4 * 32, 0, stream>>>(
      pts, 3, idxb, w1p, bn1, H, Hh, 512);

  knn_kernel<<<B2 * NPTS, 128, 0, stream>>>(H, 512, 64, idxb);
  edge_conv_kernel<64, 128, 64, 4><<<B2 * NPTS, 4 * 32, 0, stream>>>(
      H, 512, idxb, w2p, bn2, H + 64, Hh + 64, 512);

  knn_kernel<<<B2 * NPTS, 128, 0, stream>>>(H + 64, 512, 64, idxb);
  edge_conv_kernel<64, 128, 128, 8><<<B2 * NPTS, 8 * 32, 0, stream>>>(
      H + 64, 512, idxb, w3p, bn3, H + 128, Hh + 128, 512);

  knn_kernel<<<B2 * NPTS, 128, 0, stream>>>(H + 128, 512, 128, idxb);
  edge_conv_kernel<128, 256, 256, 8><<<B2 * NPTS, 8 * 32, 0, stream>>>(
      H + 128, 512, idxb, w4p, bn4, H + 256, Hh + 256, 512);

  // ---- h5 = leaky(bn(hc @ w5^T)) over all 16384 points (4 M-tiles/wave -> B reuse)
  gemm_wmma<4><<<dim3(B2 * NPTS / 64, 1024 / 16), 32, 0, stream>>>(
      Hh, 512, w5p, 512, nullptr, bn5, 1024, 512, 1, h5b, 1024, nullptr);

  reduce_h5<<<dim3(B2, 4), 256, 0, stream>>>(h5b, feat, feath);

  // ---- invariant head
  gemm_wmma<1><<<dim3(1, 1024 / 16), 32, 0, stream>>>(
      feath, 2048, ihw1p, 2048, ih_b1, ih_bn, 1024, 2048, 2, t_ih, 1024, t_h);
  gemm_wmma<1><<<dim3(1, 512 / 16), 32, 0, stream>>>(
      t_h, 1024, ihw2p, 1024, ih_b2, nullptr, 512, 1024, 0, pf, 512, nullptr);

  // ---- image branch
  gemm_wmma<1><<<dim3(80 / 16, 512 / 16), 32, 0, stream>>>(
      xmvh, 2048, imw1p, 2048, nullptr, img_bn, 512, 2048, 2, img1, 512, img1h);
  gemm_wmma<1><<<dim3(80 / 16, 512 / 16), 32, 0, stream>>>(
      img1h, 512, imw2p, 512, nullptr, nullptr, 512, 512, 0, img2, 512, nullptr);
  img_mean<<<(8 * 512 + 255) / 256, 256, 0, stream>>>(img2, imgm);

  // ---- adapters -> d_out [(8,512) x3]
  adapter_kernel<<<24, 256, 0, stream>>>(pf, imgm, ad_dw, ad_db, ad_uw, ad_ub, out);
}
